// TopKGate_63015760167573
// MI455X (gfx1250) — matrix operand order repaired and checked by reading, not compiled, not run on hardware
//
#include <hip/hip_runtime.h>

typedef __attribute__((ext_vector_type(4)))  float  v4f;
typedef __attribute__((ext_vector_type(8)))  float  v8f;
typedef __attribute__((ext_vector_type(16))) __bf16 v16bf;

#define D_MODEL   4096
#define N_EXPERTS 64
#define TOKENS    16384
#define TOK_PER_BLOCK 128
#define N_BLOCKS  (TOKENS / TOK_PER_BLOCK)

// Load 8 contiguous fp32, convert RNE to bf16, place at dst[base..base+7].
// Non-temporal variant for the streamed x matrix (read exactly once).
__device__ __forceinline__ void load8_nt(const float* __restrict__ p, v16bf& dst, int base) {
    v4f a = __builtin_nontemporal_load(reinterpret_cast<const v4f*>(p));
    v4f b = __builtin_nontemporal_load(reinterpret_cast<const v4f*>(p + 4));
    dst[base + 0] = (__bf16)a.x; dst[base + 1] = (__bf16)a.y;
    dst[base + 2] = (__bf16)a.z; dst[base + 3] = (__bf16)a.w;
    dst[base + 4] = (__bf16)b.x; dst[base + 5] = (__bf16)b.y;
    dst[base + 6] = (__bf16)b.z; dst[base + 7] = (__bf16)b.w;
}
__device__ __forceinline__ void load8_rt(const float* __restrict__ p, v16bf& dst, int base) {
    v4f a = *reinterpret_cast<const v4f*>(p);
    v4f b = *reinterpret_cast<const v4f*>(p + 4);
    dst[base + 0] = (__bf16)a.x; dst[base + 1] = (__bf16)a.y;
    dst[base + 2] = (__bf16)a.z; dst[base + 3] = (__bf16)a.w;
    dst[base + 4] = (__bf16)b.x; dst[base + 5] = (__bf16)b.y;
    dst[base + 6] = (__bf16)b.z; dst[base + 7] = (__bf16)b.w;
}

__global__ __launch_bounds__(128) void gate_kernel(const float* __restrict__ x,
                                                   const float* __restrict__ wg,
                                                   float* __restrict__ out,
                                                   float* __restrict__ ws) {
    __shared__ float s_log[TOK_PER_BLOCK * N_EXPERTS];   // logits, then probs (32 KB)
    __shared__ int   s_hard[TOK_PER_BLOCK];

    const int tid  = threadIdx.x;
    const int lane = tid & 31;
    const int wv   = tid >> 5;          // wave id 0..3
    const int r    = lane & 15;
    const int hi   = lane >> 4;         // lane half
    const int blockTok = blockIdx.x * TOK_PER_BLOCK;

    // ---------------- Phase 1: logits GEMM via bf16 WMMA ----------------
    v8f c[2][4] = {};                   // 32 tokens x 64 experts per wave

    const float* aRow[2];
    aRow[0] = x + (size_t)(blockTok + wv * 32 +  0 + r) * D_MODEL;
    aRow[1] = x + (size_t)(blockTok + wv * 32 + 16 + r) * D_MODEL;
    const float* bRow[4];
    #pragma unroll
    for (int nt = 0; nt < 4; ++nt)
        bRow[nt] = wg + (size_t)(nt * 16 + r) * D_MODEL;

    for (int k0 = 0; k0 < D_MODEL; k0 += 32) {
        v16bf a[2], b[4];
        #pragma unroll
        for (int mt = 0; mt < 2; ++mt) {
            // A (16-bit 16x32 layout): chunks at K = k0+hi*8 and K = k0+16+hi*8
            const float* p = aRow[mt] + k0 + hi * 8;
            load8_nt(p,      a[mt], 0);
            load8_nt(p + 16, a[mt], 8);
        }
        #pragma unroll
        for (int nt = 0; nt < 4; ++nt) {
            // B (32x16, lane = N): chunks at K = k0+hi*8 and K = k0+16+hi*8
            const float* p = bRow[nt] + k0 + hi * 8;
            load8_rt(p,      b[nt], 0);
            load8_rt(p + 16, b[nt], 8);
        }
        #pragma unroll
        for (int mt = 0; mt < 2; ++mt)
            #pragma unroll
            for (int nt = 0; nt < 4; ++nt)
                c[mt][nt] = __builtin_amdgcn_wmma_f32_16x16x32_bf16(
                    false, a[mt], false, b[nt], (short)0, c[mt][nt], false, false);
    }

    // Spill logits to LDS.  C layout: vgpr j, lanes 0-15 -> M=j,N=lane; 16-31 -> M=j+8,N=lane-16.
    #pragma unroll
    for (int mt = 0; mt < 2; ++mt)
        #pragma unroll
        for (int nt = 0; nt < 4; ++nt)
            #pragma unroll
            for (int j = 0; j < 8; ++j) {
                int locTok = wv * 32 + mt * 16 + hi * 8 + j;
                s_log[locTok * N_EXPERTS + nt * 16 + r] = c[mt][nt][j];
            }
    __syncthreads();

    // ---------------- Phase 2: softmax + top-2 (1 thread / token) ----------------
    {
        float* row = &s_log[tid * N_EXPERTS];
        float mx = row[0];
        #pragma unroll
        for (int e = 1; e < N_EXPERTS; ++e) mx = fmaxf(mx, row[e]);

        int i1 = 0; float l1 = row[0];
        #pragma unroll
        for (int e = 1; e < N_EXPERTS; ++e)
            if (row[e] > l1) { l1 = row[e]; i1 = e; }      // strict > : lowest index on ties
        int i2 = (i1 == 0) ? 1 : 0; float l2 = row[i2];
        #pragma unroll
        for (int e = 0; e < N_EXPERTS; ++e)
            if (e != i1 && row[e] > l2) { l2 = row[e]; i2 = e; }

        float sum = 0.0f;
        #pragma unroll
        for (int e = 0; e < N_EXPERTS; ++e) {
            float p = __expf(row[e] - mx);
            sum += p;
            row[e] = p;
        }
        float inv = 1.0f / sum;
        float v1 = row[i1] * inv, v2 = row[i2] * inv;
        #pragma unroll
        for (int e = 0; e < N_EXPERTS; ++e) row[e] *= inv;  // probs for importance

        size_t t = (size_t)blockTok + tid;
        out[t * 2 + 0] = (float)i1;                         // topk_idx (flattened, as float)
        out[t * 2 + 1] = (float)i2;
        out[(size_t)TOKENS * 2 + t * 2 + 0] = v1;           // topk_vals
        out[(size_t)TOKENS * 2 + t * 2 + 1] = v2;
        s_hard[tid] = i1;
    }
    __syncthreads();

    // ---------------- Phase 3: deterministic per-block stats ----------------
    if (tid < N_EXPERTS) {
        float imp = 0.0f; int cnt = 0;
        for (int t = 0; t < TOK_PER_BLOCK; ++t) {
            imp += s_log[t * N_EXPERTS + tid];
            cnt += (s_hard[t] == tid) ? 1 : 0;
        }
        ws[(size_t)blockIdx.x * N_EXPERTS + tid] = imp;
        ws[(size_t)N_BLOCKS * N_EXPERTS + (size_t)blockIdx.x * N_EXPERTS + tid] = (float)cnt;
    }
}

__global__ __launch_bounds__(64) void finalize_kernel(const float* __restrict__ ws,
                                                      float* __restrict__ out) {
    __shared__ float prod[N_EXPERTS];
    const int e = threadIdx.x;
    float imp = 0.0f, cnt = 0.0f;
    for (int b = 0; b < N_BLOCKS; ++b) {
        imp += ws[(size_t)b * N_EXPERTS + e];
        cnt += ws[(size_t)N_BLOCKS * N_EXPERTS + (size_t)b * N_EXPERTS + e];
    }
    prod[e] = imp * cnt;
    __syncthreads();
    if (e == 0) {
        float tot = 0.0f;
        for (int i = 0; i < N_EXPERTS; ++i) tot += prod[i];
        out[(size_t)TOKENS * 4] =
            (float)N_EXPERTS * tot / ((float)TOKENS * (float)TOKENS);
    }
}

extern "C" void kernel_launch(void* const* d_in, const int* in_sizes, int n_in,
                              void* d_out, int out_size, void* d_ws, size_t ws_size,
                              hipStream_t stream) {
    const float* x  = (const float*)d_in[0];
    const float* wg = (const float*)d_in[1];
    // d_in[2] is k == 2 (hardcoded top-2 path)
    float* out = (float*)d_out;
    float* ws  = (float*)d_ws;

    gate_kernel<<<dim3(N_BLOCKS), dim3(128), 0, stream>>>(x, wg, out, ws);
    finalize_kernel<<<dim3(1), dim3(64), 0, stream>>>(ws, out);
}
